// HGTLayerModified_49246095016487
// MI455X (gfx1250) — matrix operand order repaired and checked by reading, not compiled, not run on hardware
//
#include <hip/hip_runtime.h>
#include <math.h>

typedef __bf16 bf16_t;
typedef __attribute__((ext_vector_type(16))) __bf16 v16bf;
typedef __attribute__((ext_vector_type(8)))  float  v8f;

#define HH 8
#define FF 128
#define MB 4            // M-tiles (16 rows each) per wave in the WMMA GEMM

__device__ __forceinline__ bf16_t f2bf(float f) {
  unsigned u = __float_as_uint(f);
  unsigned r = u + 0x7FFFu + ((u >> 16) & 1u);   // round-to-nearest-even
  unsigned short h = (unsigned short)(r >> 16);
  return __builtin_bit_cast(bf16_t, h);
}

union FragU { v16bf v; unsigned u[8]; };

// ---------------- prep / elementwise kernels ----------------

__global__ void k_f32_to_bf16(const float* __restrict__ in, bf16_t* __restrict__ out, int n) {
  int i = blockIdx.x * blockDim.x + threadIdx.x;
  if (i < n) out[i] = f2bf(in[i]);
}

__global__ void k_scale_f32_to_bf16(const float* __restrict__ in, bf16_t* __restrict__ out,
                                    float s, int n) {
  int i = blockIdx.x * blockDim.x + threadIdx.x;
  if (i < n) out[i] = f2bf(in[i] * s);
}

// Build 128x128 block-diagonal (8 blocks of 16x16) from att/msg [H,DK,DK],
// stored transposed: Bt[col][k]  (col = h*16+kk, k = h*16+d)
__global__ void k_pack_blockdiag(const float* __restrict__ att, bf16_t* __restrict__ Bt) {
  int idx = blockIdx.x * blockDim.x + threadIdx.x;   // 16384
  if (idx >= FF * FF) return;
  int c = idx >> 7, i = idx & 127;
  float v = ((c >> 4) == (i >> 4)) ? att[(c >> 4) * 256 + (i & 15) * 16 + (c & 15)] : 0.f;
  Bt[idx] = f2bf(v);
}

__global__ void k_fill_f32(float* __restrict__ p, float v, int n) {
  int i = blockIdx.x * blockDim.x + threadIdx.x;
  if (i < n) p[i] = v;
}

// ---------------- WMMA GEMM: C[M,128] = A[M,128](bf16) @ Bt^T + bias ----------------
// Bt is [128,128] with Bt[col][k] = B[k][col]  (i.e. weight matrix W[out][in] as stored).
// One wave computes MB=4 vertically-stacked 16x16 tiles (64 rows), reusing each B
// fragment across 4 WMMAs; block = 8 waves covers the full N=128.

__global__ __launch_bounds__(256)
void k_gemm_bf16_wmma(const bf16_t* __restrict__ A, const bf16_t* __restrict__ Bt,
                      const float* __restrict__ bias,
                      float* __restrict__ Cf, bf16_t* __restrict__ Cb, int M)
{
  const int lane  = threadIdx.x & 31;
  const int tileN = threadIdx.x >> 5;      // 0..7
  const int half  = lane >> 4;
  const int l15   = lane & 15;
  const int baseRow = blockIdx.x * (16 * MB);
  const int bcol  = tileN * 16 + l15;

  const bf16_t* Arow[MB];
#pragma unroll
  for (int mt = 0; mt < MB; ++mt) {
    int r = baseRow + mt * 16 + l15;
    if (r >= M) r = M - 1;                 // clamp; discarded on store
    Arow[mt] = A + (size_t)r * FF;
  }
  const bf16_t* Bcol = Bt + (size_t)bcol * FF;

  v8f acc[MB];
#pragma unroll
  for (int mt = 0; mt < MB; ++mt) acc[mt] = (v8f){0.f,0.f,0.f,0.f,0.f,0.f,0.f,0.f};

#pragma unroll
  for (int k0 = 0; k0 < FF; k0 += 32) {
    // B 32x16 bf16 fragment: lane = column, same K layout as A (K base 8*half; j>=4 -> K=2j+8..)
    FragU b;
    const bf16_t* Bp = Bcol + k0 + half * 8;
#pragma unroll
    for (int j = 0; j < 8; ++j) b.u[j] = *(const unsigned*)(Bp + 2 * j + (j >= 4 ? 8 : 0));

#pragma unroll
    for (int mt = 0; mt < MB; ++mt) {
      // A 16x32 bf16 fragment (ISA layout): lanes 0-15 K base 0, lanes 16-31 K base 8;
      // VGPR i<4 -> K=2i.., VGPR i>=4 -> K=2i+8..
      FragU a;
      const bf16_t* Ap = Arow[mt] + k0 + half * 8;
#pragma unroll
      for (int i = 0; i < 4; ++i) a.u[i] = *(const unsigned*)(Ap + 2 * i);
#pragma unroll
      for (int i = 4; i < 8; ++i) a.u[i] = *(const unsigned*)(Ap + 2 * i + 8);

      acc[mt] = __builtin_amdgcn_wmma_f32_16x16x32_bf16(false, a.v, false, b.v,
                                                        (short)0, acc[mt], false, false);
    }
  }

  const float bv = bias ? bias[bcol] : 0.f;
#pragma unroll
  for (int mt = 0; mt < MB; ++mt) {
#pragma unroll
    for (int r = 0; r < 8; ++r) {
      int row = baseRow + mt * 16 + r + 8 * half;  // C layout: VGPR r -> M = r + 8*half
      if (row < M) {
        float val = acc[mt][r] + bv;
        size_t o = (size_t)row * FF + bcol;
        if (Cf) Cf[o] = val;
        if (Cb) Cb[o] = f2bf(val);
      }
    }
  }
}

// ---------------- edge-phase kernels ----------------

__device__ __forceinline__ void atomicMaxF(float* addr, float v) {
  if (v >= 0.f) atomicMax((int*)addr, __float_as_int(v));
  else          atomicMin((unsigned int*)addr, __float_as_uint(v));
}

// logits a[e,h] = dot(q[dst], ktil[src]) * pri[h] / sqrt(16); also segment-max into m
__global__ void k_edge_logits(const int* __restrict__ src, const int* __restrict__ dst,
                              const float* __restrict__ Qd, const float* __restrict__ Ktil,
                              const float* __restrict__ pri,
                              float* __restrict__ aE, float* __restrict__ m, int E)
{
  int e = blockIdx.x * blockDim.x + threadIdx.x;
  if (e >= E) return;
  int sN = src[e], dN = dst[e];
  const float4* qp = (const float4*)(Qd + (size_t)dN * FF);
  const float4* kp = (const float4*)(Ktil + (size_t)sN * FF);
#pragma unroll
  for (int h = 0; h < HH; ++h) {
    float acc = 0.f;
#pragma unroll
    for (int j = 0; j < 4; ++j) {
      float4 q = qp[h * 4 + j], k = kp[h * 4 + j];
      acc += q.x * k.x + q.y * k.y + q.z * k.z + q.w * k.w;
    }
    float aval = acc * pri[h] * 0.25f;         // 1/sqrt(DK)
    aE[(size_t)e * HH + h] = aval;
    atomicMaxF(&m[(size_t)dN * HH + h], aval);
  }
}

// e = exp(a - m[dst]); segment-sum into s
__global__ void k_edge_exp(const int* __restrict__ dst, float* __restrict__ aE,
                           const float* __restrict__ m, float* __restrict__ s, int E)
{
  int e = blockIdx.x * blockDim.x + threadIdx.x;
  if (e >= E) return;
  int dN = dst[e];
#pragma unroll
  for (int h = 0; h < HH; ++h) {
    float ev = __expf(aE[(size_t)e * HH + h] - m[(size_t)dN * HH + h]);
    aE[(size_t)e * HH + h] = ev;
    atomicAdd(&s[(size_t)dN * HH + h], ev);
  }
}

// agg[dst, h*16 + :] += (e / s[dst]) * vtil[src, h*16 + :]
__global__ void k_edge_scatter(const int* __restrict__ src, const int* __restrict__ dst,
                               const float* __restrict__ aE, const float* __restrict__ s,
                               const float* __restrict__ Vtil, float* __restrict__ agg, int E)
{
  long idx = (long)blockIdx.x * blockDim.x + threadIdx.x;
  if (idx >= (long)E * HH) return;
  int e = (int)(idx >> 3), h = (int)(idx & 7);
  int sN = src[e], dN = dst[e];
  float w = aE[(size_t)e * HH + h] / s[(size_t)dN * HH + h];
  const float* vp = Vtil + (size_t)sN * FF + h * 16;
  float* ap = agg + (size_t)dN * FF + h * 16;
#pragma unroll
  for (int i = 0; i < 16; ++i) atomicAdd(&ap[i], w * vp[i]);
}

// out = relu(trans * alpha + x * (1 - alpha)),  alpha = sigmoid(skip)
__global__ void k_final(const float* __restrict__ trans, const float* __restrict__ x,
                        const float* __restrict__ skip, float* __restrict__ out, int n)
{
  int i = blockIdx.x * blockDim.x + threadIdx.x;
  if (i >= n) return;
  float alpha = 1.f / (1.f + __expf(-skip[0]));
  float v = trans[i] * alpha + x[i] * (1.f - alpha);
  out[i] = v > 0.f ? v : 0.f;
}

// ---------------- host orchestration ----------------

extern "C" void kernel_launch(void* const* d_in, const int* in_sizes, int n_in,
                              void* d_out, int out_size, void* d_ws, size_t ws_size,
                              hipStream_t stream)
{
  (void)n_in; (void)out_size; (void)ws_size;
  const int  NT[3]     = {100000, 5000, 30000};          // word, topic, doc
  const long OFFN[3]   = {0, 100000, 105000};
  const int  NALL      = 135000;
  const int  RS[5]     = {0, 0, 0, 1, 1};                // src type per relation
  const int  RD[5]     = {0, 1, 2, 1, 2};                // dst type per relation
  const float INVCNT[3]= {1.f, 0.5f, 0.5f};              // 1 / #relations into type

  auto Pf = [&](int i) { return (const float*)d_in[i]; };
  auto Pi = [&](int i) { return (const int*)d_in[i]; };
  // d_in: 0..2 x; 3..12 (src,dst)x5; params: type t base 13+9t {Wk,bk,Wq,bq,Wv,bv,Wa,ba,skip},
  // relation r base 40+3r {pri, att, msg}

  char* ws = (char*)d_ws;
  size_t off = 0;
  auto carve = [&](size_t bytes) -> char* {
    char* p = ws + off;
    off = (off + bytes + 255) & ~(size_t)255;
    return p;
  };
  bf16_t* xb   = (bf16_t*)carve((size_t)NALL * FF * 2);   // x in bf16
  bf16_t* Kb   = (bf16_t*)carve((size_t)NALL * FF * 2);   // K projection (bf16, GEMM input)
  bf16_t* Vb   = (bf16_t*)carve((size_t)NALL * FF * 2);   // V projection (bf16)
  float*  Qf   = (float*) carve((size_t)NALL * FF * 4);   // Q projection (f32)
  float*  Ktil = (float*) carve((size_t)100000 * FF * 4); // per-relation K~ (reused)
  float*  Vtil = (float*) carve((size_t)100000 * FF * 4); // per-relation V~ (reused)
  float*  aE   = (float*) carve((size_t)400000 * HH * 4); // edge logits/weights (reused)
  float*  mB   = (float*) carve((size_t)100000 * HH * 4); // segment max (reused)
  float*  sB   = (float*) carve((size_t)100000 * HH * 4); // segment sum (reused)
  float*  agg  = (float*) carve((size_t)NALL * FF * 4);   // aggregated messages
  bf16_t* tfb  = (bf16_t*)carve((size_t)100000 * FF * 2); // tf in bf16 (reused)
  bf16_t* Wb   = (bf16_t*)carve((size_t)12 * FF * FF * 2);// 12 weight mats bf16
  bf16_t* BD   = (bf16_t*)carve((size_t)10 * FF * FF * 2);// 10 block-diag att/msg bf16

  const int TPB = 256;
  auto blocks = [&](long n) { return (unsigned)((n + TPB - 1) / TPB); };
  auto gtiles = [&](int M)  { return (unsigned)((M + 16 * MB - 1) / (16 * MB)); };

  // 1. x -> bf16
  for (int t = 0; t < 3; ++t) {
    int n = NT[t] * FF;
    k_f32_to_bf16<<<blocks(n), TPB, 0, stream>>>(Pf(t), xb + OFFN[t] * FF, n);
  }
  // 2. weights -> bf16 (Bt layout == W row-major as stored)
  for (int t = 0; t < 3; ++t)
    for (int p = 0; p < 4; ++p)
      k_f32_to_bf16<<<blocks(FF * FF), TPB, 0, stream>>>(
          Pf(13 + 9 * t + 2 * p), Wb + (size_t)(t * 4 + p) * FF * FF, FF * FF);
  // 3. att/msg -> block-diagonal bf16
  for (int r = 0; r < 5; ++r) {
    k_pack_blockdiag<<<blocks(FF * FF), TPB, 0, stream>>>(Pf(40 + 3 * r + 1),
                                                          BD + (size_t)(2 * r) * FF * FF);
    k_pack_blockdiag<<<blocks(FF * FF), TPB, 0, stream>>>(Pf(40 + 3 * r + 2),
                                                          BD + (size_t)(2 * r + 1) * FF * FF);
  }
  // 4. zero agg
  k_fill_f32<<<blocks((long)NALL * FF), TPB, 0, stream>>>(agg, 0.f, NALL * FF);

  // 5. K/Q/V projections (WMMA GEMMs)
  for (int t = 0; t < 3; ++t) {
    const bf16_t* A = xb + OFFN[t] * FF;
    k_gemm_bf16_wmma<<<gtiles(NT[t]), 256, 0, stream>>>(A, Wb + (size_t)(t * 4 + 0) * FF * FF,
                                                        Pf(13 + 9 * t + 1), nullptr,
                                                        Kb + OFFN[t] * FF, NT[t]);
    k_gemm_bf16_wmma<<<gtiles(NT[t]), 256, 0, stream>>>(A, Wb + (size_t)(t * 4 + 1) * FF * FF,
                                                        Pf(13 + 9 * t + 3), Qf + OFFN[t] * FF,
                                                        nullptr, NT[t]);
    k_gemm_bf16_wmma<<<gtiles(NT[t]), 256, 0, stream>>>(A, Wb + (size_t)(t * 4 + 2) * FF * FF,
                                                        Pf(13 + 9 * t + 5), nullptr,
                                                        Vb + OFFN[t] * FF, NT[t]);
  }

  // 6. relations: node-level att/msg transforms (WMMA) + edge softmax + scatter
  for (int r = 0; r < 5; ++r) {
    int st = RS[r], dt = RD[r];
    int E = in_sizes[3 + 2 * r];
    const int* src = Pi(3 + 2 * r);
    const int* dst = Pi(4 + 2 * r);
    k_gemm_bf16_wmma<<<gtiles(NT[st]), 256, 0, stream>>>(Kb + OFFN[st] * FF,
                                                         BD + (size_t)(2 * r) * FF * FF,
                                                         nullptr, Ktil, nullptr, NT[st]);
    k_gemm_bf16_wmma<<<gtiles(NT[st]), 256, 0, stream>>>(Vb + OFFN[st] * FF,
                                                         BD + (size_t)(2 * r + 1) * FF * FF,
                                                         nullptr, Vtil, nullptr, NT[st]);
    int nd = NT[dt] * HH;
    k_fill_f32<<<blocks(nd), TPB, 0, stream>>>(mB, -INFINITY, nd);
    k_fill_f32<<<blocks(nd), TPB, 0, stream>>>(sB, 0.f, nd);
    k_edge_logits<<<blocks(E), TPB, 0, stream>>>(src, dst, Qf + OFFN[dt] * FF, Ktil,
                                                 Pf(40 + 3 * r), aE, mB, E);
    k_edge_exp<<<blocks(E), TPB, 0, stream>>>(dst, aE, mB, sB, E);
    k_edge_scatter<<<blocks((long)E * HH), TPB, 0, stream>>>(src, dst, aE, sB, Vtil,
                                                             agg + OFFN[dt] * FF, E);
  }

  // 7. output projection (WMMA) + gated skip + relu
  float* trans = Ktil;  // reuse (edge phase complete)
  for (int t = 0; t < 3; ++t) {
    int n = NT[t] * FF;
    k_scale_f32_to_bf16<<<blocks(n), TPB, 0, stream>>>(agg + OFFN[t] * FF, tfb, INVCNT[t], n);
    k_gemm_bf16_wmma<<<gtiles(NT[t]), 256, 0, stream>>>(tfb, Wb + (size_t)(t * 4 + 3) * FF * FF,
                                                        Pf(13 + 9 * t + 7), trans, nullptr, NT[t]);
    k_final<<<blocks(n), TPB, 0, stream>>>(trans, Pf(t), Pf(13 + 9 * t + 8),
                                           (float*)d_out + OFFN[t] * FF, n);
  }
}
